// DLSTM_86938728006339
// MI455X (gfx1250) — matrix operand order repaired and checked by reading, not compiled, not run on hardware
//
#include <hip/hip_runtime.h>
#include <stdint.h>

// ---------------- problem dims ----------------
#define D_SEQ 1024
#define D_B   64
#define D_IN  512
#define D_H   512
#define D_XC  1024          // H + IN
#define D_Y   2048          // 4*H
#define NWG   16
#define TPB   256
#define OUT_STEP (D_B * D_H)   // 32768 elements per step

typedef __attribute__((ext_vector_type(16))) _Float16 v16h;
typedef __attribute__((ext_vector_type(8)))  _Float16 v8h;
typedef __attribute__((ext_vector_type(8)))  float    v8f;
typedef __attribute__((ext_vector_type(4)))  unsigned u32x4;
typedef __attribute__((ext_vector_type(8)))  int      i32x8;
typedef __attribute__((ext_vector_type(4)))  int      i32x4;

// ---------------- fragment loaders (ISA 7.12.2 layouts, wave32) ----------------
__device__ __forceinline__ v16h frag_combine(v8h lo, v8h hi) {
  return __builtin_shufflevector(lo, hi, 0,1,2,3,4,5,6,7,8,9,10,11,12,13,14,15);
}

// A 16x32 f16: lane<16 row M=lane, K chunks {0..7},{16..23}; lane>=16 same row, {8..15},{24..31}
__device__ __forceinline__ v16h loadA(const _Float16* rowBase, int laneHi) {
  v8h lo = *(const v8h*)(rowBase + laneHi * 8);
  v8h hi = *(const v8h*)(rowBase + 16 + laneHi * 8);
  return frag_combine(lo, hi);
}
// B 32x16 f16: lane holds column N=lane%16, K chunks {0..7},{16..23} (lane<16); {8..15},{24..31} (lane>=16)
__device__ __forceinline__ v16h loadB(const _Float16* colBase, int laneHi) {
  const _Float16* p = colBase + laneHi * 8;
  v8h lo = *(const v8h*)(p);
  v8h hi = *(const v8h*)(p + 16);
  return frag_combine(lo, hi);
}

__device__ __forceinline__ v8f wmma_f16(v16h a, v16h b, v8f c) {
  return __builtin_amdgcn_wmma_f32_16x16x32_f16(false, a, false, b, (short)0, c, false, false);
}

// ---------------- device-wide barrier (persistent kernel) ----------------
__device__ __forceinline__ void gridBarrier(int* bar) {
  __threadfence();
  __syncthreads();
  if (threadIdx.x == 0) {
    int gen  = __hip_atomic_load(&bar[1], __ATOMIC_RELAXED, __HIP_MEMORY_SCOPE_AGENT);
    int prev = __hip_atomic_fetch_add(&bar[0], 1, __ATOMIC_ACQ_REL, __HIP_MEMORY_SCOPE_AGENT);
    if (prev == NWG - 1) {
      __hip_atomic_store(&bar[0], 0, __ATOMIC_RELAXED, __HIP_MEMORY_SCOPE_AGENT);
      __hip_atomic_store(&bar[1], gen + 1, __ATOMIC_RELEASE, __HIP_MEMORY_SCOPE_AGENT);
    } else {
      while (__hip_atomic_load(&bar[1], __ATOMIC_ACQUIRE, __HIP_MEMORY_SCOPE_AGENT) == gen) {
        __builtin_amdgcn_s_sleep(1);
      }
    }
  }
  __syncthreads();
  __threadfence();
}

// ---------------- TDM: async tensor load of a contiguous block into LDS ----------------
#if __has_builtin(__builtin_amdgcn_tensor_load_to_lds)
#define HAVE_TDM 1
__device__ __forceinline__ void tdm_load_1d(const void* gsrc, void* ldst, unsigned bytes) {
  unsigned units = bytes >> 3;                       // data_size = 8 bytes
  unsigned long long ga = (unsigned long long)(uintptr_t)gsrc;
  unsigned lds = (unsigned)(uintptr_t)ldst;          // low 32 bits of generic addr = LDS offset
  // D# group0: count=1 | lds_addr | global_addr[56:0] | type=2
  u32x4 g0 = { 1u, lds, (unsigned)ga,
               0x80000000u | (unsigned)((ga >> 32) & 0x1FFFFFFull) };
  // D# group1: data_size=3 (8B); tensor_dim0 = tile_dim0 = units; dim1 = 1; stride0 = units
  i32x8 g1 = { (int)0x00030000u,
               (int)((units & 0xFFFFu) << 16),        // tensor_dim0 lo16 in [63:48]
               (int)(((units >> 16) & 0xFFFFu) | (1u << 16)), // td0 hi16 | tensor_dim1=1
               (int)((units & 0xFFFFu) << 16),        // tile_dim0 in [127:112]
               1,                                     // tile_dim1 = 1
               (int)units,                            // tensor_dim0_stride lo32
               0, 0 };
  i32x4 g2 = { 1, 1, 0, 0 };                          // tensor_dim2=1, tensor_dim3=1
  i32x4 g3 = { 0, 0x00010000, 0, 0 };                 // tensor_dim4=1
  i32x8 g4 = { 0, 0, 0, 0, 0, 0, 0, 0 };              // trailing group (clang-23 6-arg form)
  __builtin_amdgcn_tensor_load_to_lds(g0, g1, g2, g3, g4, 0);
}
#endif

// ---------------- prep: f32->f16 weights, init xc/c/barrier ----------------
__global__ void dlstm_prep(const float* __restrict__ inp,
                           const float* __restrict__ W0,
                           const float* __restrict__ W1,
                           _Float16* __restrict__ W0h, _Float16* __restrict__ W1h,
                           _Float16* __restrict__ xc, float* __restrict__ cbuf,
                           int* __restrict__ bar) {
  int i = blockIdx.x * TPB + threadIdx.x;
  if (i < D_H * D_XC)  W0h[i] = (_Float16)W0[i];
  if (i < D_Y * D_H)   W1h[i] = (_Float16)W1[i];
  if (i < D_B * D_H) {
    cbuf[i] = 0.f;
    int m = i >> 9, j = i & 511;
    xc[m * D_XC + j]        = (_Float16)inp[i];   // x_0
    xc[m * D_XC + D_H + j]  = (_Float16)0.f;      // c_0 = 0
  }
  if (i < 8) bar[i] = 0;
}

// ---------------- persistent main kernel ----------------
__global__ void __launch_bounds__(TPB, 1)
dlstm_main(const float* __restrict__ inp,
           const float* __restrict__ b0, const float* __restrict__ b1,
           const _Float16* __restrict__ W0h, const _Float16* __restrict__ W1h,
           _Float16* __restrict__ xc, _Float16* __restrict__ hbuf,
           float* __restrict__ ybuf, float* __restrict__ cbuf,
           int* __restrict__ bar, float* __restrict__ out) {
  extern __shared__ _Float16 smem[];
  _Float16* sW0 = smem;                 // [32][1024]  = 64 KB (this WG's W0 columns)
  _Float16* sW1 = smem + 32 * D_XC;     // [128][512]  = 128 KB (this WG's W1 rows)

  const int wg   = blockIdx.x;
  const int tid  = threadIdx.x;
  const int wave = tid >> 5;
  const int lane = tid & 31;
  const int laneHi = lane >> 4;
  const int ln   = lane & 15;

  // ---- preload weight slices into LDS via Tensor Data Mover ----
#if defined(HAVE_TDM)
  if (tid < 32) {
    tdm_load_1d(W0h + (size_t)wg * 32 * D_XC, sW0, 32 * D_XC * 2);
    tdm_load_1d(W1h + (size_t)wg * 128 * D_H, sW1, 128 * D_H * 2);
#if __has_builtin(__builtin_amdgcn_s_wait_tensorcnt)
    __builtin_amdgcn_s_wait_tensorcnt(0);
#else
    asm volatile("s_wait_tensorcnt 0x0" ::: "memory");
#endif
  }
#else
  {
    const v8h* s0 = (const v8h*)(W0h + (size_t)wg * 32 * D_XC);
    v8h* d0 = (v8h*)sW0;
    for (int i = tid; i < 32 * D_XC / 8; i += TPB) d0[i] = s0[i];
    const v8h* s1 = (const v8h*)(W1h + (size_t)wg * 128 * D_H);
    v8h* d1 = (v8h*)sW1;
    for (int i = tid; i < 128 * D_H / 8; i += TPB) d1[i] = s1[i];
  }
#endif
  __syncthreads();

  for (int t = 0; t < D_SEQ; ++t) {
    // ---- phase B: h = relu(xc @ W0^T + b0), one 16x16 tile per wave ----
    {
      const int mt  = wave & 3;          // 4 row tiles (M = 64)
      const int ntl = wave >> 2;         // 2 local column tiles
      const _Float16* aRow = xc + (mt * 16 + ln) * D_XC;
      const _Float16* bCol = sW0 + (ntl * 16 + ln) * D_XC;
      v8f acc = {};
      for (int k0 = 0; k0 < D_XC; k0 += 32) {
        v16h a = loadA(aRow + k0, laneHi);
        v16h b = loadB(bCol + k0, laneHi);
        acc = wmma_f16(a, b, acc);
      }
      const int n = (wg * 2 + ntl) * 16 + ln;
      const float bias = b0[n];
#pragma unroll
      for (int r = 0; r < 8; ++r) {
        int m = mt * 16 + r + laneHi * 8;
        float v = acc[r] + bias;
        hbuf[m * D_H + n] = (_Float16)(v > 0.f ? v : 0.f);
      }
      if (t + 1 < D_SEQ)   // warm L2 for next step's input slice
        __builtin_prefetch(inp + ((size_t)(t + 1) << 15) + (wg * TPB + tid) * 8, 0, 0);
    }
    gridBarrier(bar);

    // ---- phase C: y = h @ W1^T + b1 ----
    // k-major with 4 live accumulators: A fragment loaded once per k-step and
    // reused by 4 WMMAs (4 independent D-chains also hide XDL latency).
    {
      const int mt      = wave & 3;
      const int ntlBase = (wave >> 2) * 4;          // 4 local col tiles per wave
      const _Float16* aRow = hbuf + (mt * 16 + ln) * D_H;
      const _Float16* bCol[4];
#pragma unroll
      for (int q = 0; q < 4; ++q)
        bCol[q] = sW1 + ((ntlBase + q) * 16 + ln) * D_H;
      v8f acc[4] = {};
      for (int k0 = 0; k0 < D_H; k0 += 32) {
        v16h a = loadA(aRow + k0, laneHi);
#pragma unroll
        for (int q = 0; q < 4; ++q)
          acc[q] = wmma_f16(a, loadB(bCol[q] + k0, laneHi), acc[q]);
      }
#pragma unroll
      for (int q = 0; q < 4; ++q) {
        const int n = wg * 128 + (ntlBase + q) * 16 + ln;
        const float bias = b1[n];
#pragma unroll
        for (int r = 0; r < 8; ++r) {
          int m = mt * 16 + r + laneHi * 8;
          ybuf[m * D_Y + n] = acc[q][r] + bias;
        }
      }
    }
    gridBarrier(bar);

    // ---- phase D: gating + state update + output + next xc ----
    {
      const int gtid = wg * TPB + tid;
      for (int i = gtid; i < D_B * D_H; i += NWG * TPB) {
        const int m = i >> 9, j = i & 511;
        const float* yr = ybuf + m * D_Y;
        float ya = yr[j], yb2 = yr[D_H + j], hx = yr[2 * D_H + j], fg = yr[3 * D_H + j];
        float ncx = (ya > 0.f ? ya : 0.f) - (yb2 > 0.f ? yb2 : 0.f);   // drelu
        float nhx = hx > 0.f ? hx : 0.f;
        float f   = 1.f / (1.f + __expf(-fg));                          // forget
        float cn  = f * ncx + (1.f - f) * cbuf[i];
        cbuf[i] = cn;
        out[(size_t)t * OUT_STEP + i] = nhx;
        xc[m * D_XC + D_H + j] = (_Float16)cn;                          // c for next step
        if (t + 1 < D_SEQ)
          xc[m * D_XC + j] = (_Float16)inp[((size_t)(t + 1) << 15) + i]; // x_{t+1}
        else
          out[(size_t)D_SEQ * OUT_STEP + i] = cn;                       // c_final tail
      }
    }
    gridBarrier(bar);
  }
}

// ---------------- host entry ----------------
extern "C" void kernel_launch(void* const* d_in, const int* in_sizes, int n_in,
                              void* d_out, int out_size, void* d_ws, size_t ws_size,
                              hipStream_t stream) {
  const float* inp = (const float*)d_in[0];
  const float* W0  = (const float*)d_in[1];
  const float* b0  = (const float*)d_in[2];
  const float* W1  = (const float*)d_in[3];
  const float* b1  = (const float*)d_in[4];

  char* ws = (char*)d_ws;
  _Float16* W0h = (_Float16*)(ws + 0);          // 512*1024*2   = 1048576
  _Float16* W1h = (_Float16*)(ws + 1048576);    // 2048*512*2   = 2097152
  _Float16* xcb = (_Float16*)(ws + 3145728);    // 64*1024*2    = 131072
  _Float16* hb  = (_Float16*)(ws + 3276800);    // 64*512*2     = 65536
  float*    yb  = (float*)   (ws + 3342336);    // 64*2048*4    = 524288
  float*    cb  = (float*)   (ws + 3866624);    // 64*512*4     = 131072
  int*      bar = (int*)     (ws + 3997696);    // barrier state

  dlstm_prep<<<(D_Y * D_H + TPB - 1) / TPB, TPB, 0, stream>>>(
      inp, W0, W1, W0h, W1h, xcb, cb, bar);

  const size_t ldsBytes = (size_t)(32 * D_XC + 128 * D_H) * sizeof(_Float16); // 192 KB
  dlstm_main<<<NWG, TPB, ldsBytes, stream>>>(
      inp, b0, b1, W0h, W1h, xcb, hb, yb, cb, bar, (float*)d_out);
}
